// TransformerEncoderLayer_30889404793579
// MI455X (gfx1250) — compile-verified, matrix operand order checked by reading, not run on hardware
//
#include <hip/hip_runtime.h>
#include <hip/hip_bf16.h>
#include <stdint.h>

// ---------------------------------------------------------------------------
// Types
// ---------------------------------------------------------------------------
typedef __bf16 bf16;
typedef __attribute__((ext_vector_type(16))) __bf16 v16bf;
typedef __attribute__((ext_vector_type(8)))  __bf16 v8bf;
typedef __attribute__((ext_vector_type(4)))  __bf16 v4bf;
typedef __attribute__((ext_vector_type(8)))  float  v8f;

#define SEQ 2048
#define DMODEL 1024
#define NHEAD 16
#define HD 64
#define FFDIM 4096

// ---------------------------------------------------------------------------
// CDNA5 async global->LDS copies (ASYNCcnt-tracked). The LDS address operand
// is the workgroup-relative byte offset; for a generic pointer to __shared__
// memory that is the low 32 bits of the address (ISA 10.2 aperture mapping:
// LDS offset is the low dword of the address).
// ---------------------------------------------------------------------------
__device__ __forceinline__ uint32_t lds_off_u32(const void* p) {
  return (uint32_t)(uintptr_t)p;
}
__device__ __forceinline__ void async_copy_b128(void* lds, const void* gaddr) {
  asm volatile("global_load_async_to_lds_b128 %0, %1, off"
               :: "v"(lds_off_u32(lds)), "v"(gaddr) : "memory");
}
__device__ __forceinline__ void async_copy_b32(void* lds, const void* gaddr) {
  asm volatile("global_load_async_to_lds_b32 %0, %1, off"
               :: "v"(lds_off_u32(lds)), "v"(gaddr) : "memory");
}
__device__ __forceinline__ void wait_async() {
  asm volatile("s_wait_asynccnt 0x0" ::: "memory");
}

// ---------------------------------------------------------------------------
// WMMA helpers (CDNA5 wave32: D = A(16x32 bf16) * B(32x16 bf16) + C(16x16 f32))
// ---------------------------------------------------------------------------
__device__ __forceinline__ v8f wmma_bf16(v16bf a, v16bf b, v8f c) {
  return __builtin_amdgcn_wmma_f32_16x16x32_bf16(
      /*neg_a=*/false, a, /*neg_b=*/false, b,
      /*c_mod=*/(short)0, c, /*reuse_a=*/false, /*reuse_b=*/false);
}

// 16x32 bf16 fragment from LDS where the 32 K-values of each row/col are
// contiguous with stride `ld` (ld % 8 == 0 => both 16B chunks aligned).
// ISA 7.12.2: lane l -> m = l&15, g = l>>4; half i: k = (i<8?0:16)+g*8+(i&7).
__device__ __forceinline__ v16bf load_frag(const bf16* p0, int ld, int lane) {
  const bf16* p = p0 + (size_t)(lane & 15) * ld + ((lane >> 4) << 3);
  v8bf lo = *(const v8bf*)(p);
  v8bf hi = *(const v8bf*)(p + 16);
  v16bf r;
#pragma unroll
  for (int i = 0; i < 8; ++i) { r[i] = lo[i]; r[8 + i] = hi[i]; }
  return r;
}

// ---------------------------------------------------------------------------
// Async tile staging: A row-major [M][K], Bt N-major [N][K]; both tiles land
// K-contiguous in LDS with stride 40 halves.
// ---------------------------------------------------------------------------
template <int BM, int BN>
__device__ __forceinline__ void stage_async(
    const bf16* __restrict__ A, const bf16* __restrict__ Bt,
    int lda, int ldbt, int row0, int col0, int kb,
    bf16* __restrict__ As, bf16* __restrict__ Bs, int tid) {
#pragma unroll
  for (int c = 0; c < BM * 4 / 256; ++c) {
    int cc = tid + c * 256;
    int r = cc >> 2, ks = (cc & 3) << 3;
    async_copy_b128(&As[r * 40 + ks], A + (size_t)(row0 + r) * lda + kb + ks);
  }
#pragma unroll
  for (int c = 0; c < BN * 4 / 256; ++c) {
    int cc = tid + c * 256;
    int n = cc >> 2, ks = (cc & 3) << 3;
    async_copy_b128(&Bs[n * 40 + ks], Bt + (size_t)(col0 + n) * ldbt + kb + ks);
  }
}

// ---------------------------------------------------------------------------
// Generic bf16 GEMM core: C[M,N] = act(A[M,K] @ Bt[N,K]^T + bias)
// Double-buffered LDS; async copies overlap the WMMA stream. 8 waves, BK=32.
// ---------------------------------------------------------------------------
template <int BM, int BN, int WAVES_M, int WAVES_N, bool RELU, bool OUT_BF16,
          bool OUT_TRANS>
__device__ __forceinline__ void gemm_core(
    const bf16* __restrict__ A, const bf16* __restrict__ Bt,
    const float* __restrict__ bias, void* __restrict__ Cv,
    int K, int lda, int ldbt, int ldc, int row0, int col0) {
  constexpr int TM = BM / (WAVES_M * 16);
  constexpr int TN = BN / (WAVES_N * 16);
  __shared__ bf16 As[2][BM][40];
  __shared__ bf16 Bs[2][BN][40];

  const int tid = threadIdx.x;
  const int lane = tid & 31;
  const int w = tid >> 5;
  const int wm = w / WAVES_N;
  const int wn = w % WAVES_N;

  v8f acc[TM][TN] = {};

  stage_async<BM, BN>(A, Bt, lda, ldbt, row0, col0, 0, &As[0][0][0],
                      &Bs[0][0][0], tid);
  wait_async();
  __syncthreads();

  int buf = 0;
  for (int kb = 0; kb < K; kb += 32) {
    const bool more = (kb + 32) < K;
    if (more)  // producer: async-fill the other buffer while we compute
      stage_async<BM, BN>(A, Bt, lda, ldbt, row0, col0, kb + 32,
                          &As[buf ^ 1][0][0], &Bs[buf ^ 1][0][0], tid);

    v16bf af[TM], bfr[TN];
#pragma unroll
    for (int i = 0; i < TM; ++i)
      af[i] = load_frag(&As[buf][wm * TM * 16 + i * 16][0], 40, lane);
#pragma unroll
    for (int j = 0; j < TN; ++j)
      bfr[j] = load_frag(&Bs[buf][wn * TN * 16 + j * 16][0], 40, lane);
#pragma unroll
    for (int i = 0; i < TM; ++i)
#pragma unroll
      for (int j = 0; j < TN; ++j)
        acc[i][j] = wmma_bf16(af[i], bfr[j], acc[i][j]);

    if (more) wait_async();
    __syncthreads();
    buf ^= 1;
  }

  // epilogue: C layout lane mapping: n = lane&15, m = r + 8*(lane>>4)
  const int r0 = row0 + wm * TM * 16;
  const int c0 = col0 + wn * TN * 16;
#pragma unroll
  for (int i = 0; i < TM; ++i)
#pragma unroll
    for (int j = 0; j < TN; ++j)
#pragma unroll
      for (int r = 0; r < 8; ++r) {
        int row = r0 + i * 16 + ((lane >> 4) << 3) + r;
        int col = c0 + j * 16 + (lane & 15);
        float v = acc[i][j][r] + bias[col];
        if (RELU) v = fmaxf(v, 0.0f);
        if (OUT_TRANS)
          ((bf16*)Cv)[(size_t)col * ldc + row] = (bf16)v;
        else if (OUT_BF16)
          ((bf16*)Cv)[(size_t)row * ldc + col] = (bf16)v;
        else
          ((float*)Cv)[(size_t)row * ldc + col] = v;
      }
}

// ---------------------------------------------------------------------------
// Conversion kernels
// ---------------------------------------------------------------------------
__global__ void k_cvt_bf16(const float* __restrict__ in, bf16* __restrict__ out,
                           int n) {
  int i = (blockIdx.x * 256 + threadIdx.x) * 4;
  if (i + 3 < n) {
    float4 f = *(const float4*)(in + i);
    v4bf o;
    o[0] = (bf16)f.x; o[1] = (bf16)f.y; o[2] = (bf16)f.z; o[3] = (bf16)f.w;
    *(v4bf*)(out + i) = o;
  }
}

// Transpose+convert: out[c*R + r] = (bf16)in[r*C + c], batched over blockIdx.z.
__global__ void __launch_bounds__(256) k_cvt_t(const float* __restrict__ in,
                                               bf16* __restrict__ out, int R,
                                               int C) {
  __shared__ float t[32][33];
  const float* I = in + (size_t)blockIdx.z * R * C;
  bf16* O = out + (size_t)blockIdx.z * R * C;
  int r0 = blockIdx.y * 32, c0 = blockIdx.x * 32;
  int tx = threadIdx.x & 31, ty = threadIdx.x >> 5;  // 32 x 8
#pragma unroll
  for (int i = 0; i < 32; i += 8)
    t[ty + i][tx] = I[(size_t)(r0 + ty + i) * C + c0 + tx];
  __syncthreads();
#pragma unroll
  for (int i = 0; i < 32; i += 8)
    O[(size_t)(c0 + ty + i) * R + r0 + tx] = (bf16)t[tx][ty + i];
}

__global__ void k_zero_f32(float* __restrict__ p, int n) {
  int i = blockIdx.x * 256 + threadIdx.x;
  if (i < n) p[i] = 0.0f;
}

// ---------------------------------------------------------------------------
// QKV projections, one head + one of {Q,K,V} per blockIdx.z.
// Weights pre-transposed: Wt[h] is [HD][DMODEL]. V output stored transposed
// (Vt[h][e][s]) so attention can stage it with contiguous async copies.
// ---------------------------------------------------------------------------
__global__ void __launch_bounds__(256) k_qkv(
    const bf16* __restrict__ xb, const bf16* __restrict__ Wqt,
    const bf16* __restrict__ Wkt, const bf16* __restrict__ Wvt,
    const float* __restrict__ bq, const float* __restrict__ bk,
    const float* __restrict__ bv, bf16* __restrict__ Qb,
    bf16* __restrict__ Kb, bf16* __restrict__ Vt) {
  int z = blockIdx.z;
  int h = z & 15;
  int which = z >> 4;
  int row0 = blockIdx.x * 128;
  if (which == 0) {
    gemm_core<128, 64, 8, 1, false, true, false>(
        xb, Wqt + (size_t)h * HD * DMODEL, bq + h * HD,
        Qb + (size_t)h * SEQ * HD, DMODEL, DMODEL, DMODEL, HD, row0, 0);
  } else if (which == 1) {
    gemm_core<128, 64, 8, 1, false, true, false>(
        xb, Wkt + (size_t)h * HD * DMODEL, bk + h * HD,
        Kb + (size_t)h * SEQ * HD, DMODEL, DMODEL, DMODEL, HD, row0, 0);
  } else {
    gemm_core<128, 64, 8, 1, false, true, true>(
        xb, Wvt + (size_t)h * HD * DMODEL, bv + h * HD,
        Vt + (size_t)h * HD * SEQ, DMODEL, DMODEL, DMODEL, SEQ, row0, 0);
  }
}

// ---------------------------------------------------------------------------
// Attention pass 1: Z[h][s] = sum_t exp(q_s . k_t / 8)
// ---------------------------------------------------------------------------
__global__ void __launch_bounds__(256) k_attn_z(const bf16* __restrict__ Qb,
                                                const bf16* __restrict__ Kb,
                                                float* __restrict__ Z) {
  __shared__ bf16 Qs[128][72];
  __shared__ bf16 Ks[128][72];
  const int h = blockIdx.z;
  const int s0 = blockIdx.x * 128;
  const int tid = threadIdx.x, lane = tid & 31, w = tid >> 5;
  const int wm = w >> 1, wn = w & 1;  // 4x2 waves, wave tile = 32s x 64t
  const bf16* Qh = Qb + (size_t)h * SEQ * HD;
  const bf16* Kh = Kb + (size_t)h * SEQ * HD;

#pragma unroll
  for (int c = 0; c < 4; ++c) {  // 128x64, 8-half chunks
    int cc = tid + c * 256;
    int r = cc >> 3, ks = (cc & 7) << 3;
    async_copy_b128(&Qs[r][ks], Qh + (size_t)(s0 + r) * HD + ks);
  }
  wait_async();
  __syncthreads();

  v16bf qf[2][2];
#pragma unroll
  for (int ks2 = 0; ks2 < 2; ++ks2)
#pragma unroll
    for (int i = 0; i < 2; ++i)
      qf[ks2][i] = load_frag(&Qs[wm * 32 + i * 16][ks2 * 32], 72, lane);

  for (int t0 = 0; t0 < SEQ; t0 += 128) {
    __syncthreads();
#pragma unroll
    for (int c = 0; c < 4; ++c) {
      int cc = tid + c * 256;
      int r = cc >> 3, ks = (cc & 7) << 3;
      async_copy_b128(&Ks[r][ks], Kh + (size_t)(t0 + r) * HD + ks);
    }
    wait_async();
    __syncthreads();

    v8f acc[2][4] = {};
#pragma unroll
    for (int ks2 = 0; ks2 < 2; ++ks2)
#pragma unroll
      for (int j = 0; j < 4; ++j) {
        v16bf b = load_frag(&Ks[wn * 64 + j * 16][ks2 * 32], 72, lane);
#pragma unroll
        for (int i = 0; i < 2; ++i)
          acc[i][j] = wmma_bf16(qf[ks2][i], b, acc[i][j]);
      }

#pragma unroll
    for (int i = 0; i < 2; ++i)
#pragma unroll
      for (int r = 0; r < 8; ++r) {
        float p = 0.0f;
#pragma unroll
        for (int j = 0; j < 4; ++j) p += __expf(acc[i][j][r] * 0.125f);
        p += __shfl_xor(p, 1, 32);
        p += __shfl_xor(p, 2, 32);
        p += __shfl_xor(p, 4, 32);
        p += __shfl_xor(p, 8, 32);
        if ((lane & 15) == 0) {
          int s = s0 + wm * 32 + i * 16 + ((lane >> 4) << 3) + r;
          atomicAdd(&Z[(size_t)h * SEQ + s], p);
        }
      }
  }
}

// ---------------------------------------------------------------------------
// Attention pass 2: out[t,e] = sum_s exp(k_t . q_s / 8) / Z_s * V[s,e]
// V consumed pre-transposed (Vt[h][e][s]); result -> concat[t, h*64+e] bf16.
// ---------------------------------------------------------------------------
__global__ void __launch_bounds__(256) k_attn_out(
    const bf16* __restrict__ Qb, const bf16* __restrict__ Kb,
    const bf16* __restrict__ Vt, const float* __restrict__ Z,
    bf16* __restrict__ concat) {
  __shared__ bf16 Kt[128][72];    // A for scores: rows t
  __shared__ bf16 Qs[128][72];    // B for scores: rows s (n-major)
  __shared__ bf16 Ps[128][136];   // P~[t][s] tile, bf16
  __shared__ bf16 Vs[64][136];    // B for PV: Vs[e][s]
  __shared__ float Zs[128];
  const int h = blockIdx.z;
  const int t0 = blockIdx.x * 128;
  const int tid = threadIdx.x, lane = tid & 31, w = tid >> 5;
  const bf16* Qh = Qb + (size_t)h * SEQ * HD;
  const bf16* Kh = Kb + (size_t)h * SEQ * HD;
  const bf16* Vth = Vt + (size_t)h * HD * SEQ;

#pragma unroll
  for (int c = 0; c < 4; ++c) {
    int cc = tid + c * 256;
    int r = cc >> 3, ks = (cc & 7) << 3;
    async_copy_b128(&Kt[r][ks], Kh + (size_t)(t0 + r) * HD + ks);
  }
  wait_async();
  __syncthreads();

  v16bf ktf[2];  // this wave's 16 t-rows, two K-steps
  ktf[0] = load_frag(&Kt[w * 16][0], 72, lane);
  ktf[1] = load_frag(&Kt[w * 16][32], 72, lane);

  v8f oacc[4] = {};  // 16t x 64e per wave

  for (int s0 = 0; s0 < SEQ; s0 += 128) {
    __syncthreads();
#pragma unroll
    for (int c = 0; c < 4; ++c) {  // Qs rows (contiguous)
      int cc = tid + c * 256;
      int r = cc >> 3, ks = (cc & 7) << 3;
      async_copy_b128(&Qs[r][ks], Qh + (size_t)(s0 + r) * HD + ks);
    }
#pragma unroll
    for (int c = 0; c < 4; ++c) {  // Vs rows from Vt (contiguous!)
      int cc = tid + c * 256;
      int e = cc >> 4, ks = (cc & 15) << 3;
      async_copy_b128(&Vs[e][ks], Vth + (size_t)e * SEQ + s0 + ks);
    }
    if (tid < 128) async_copy_b32(&Zs[tid], &Z[(size_t)h * SEQ + s0 + tid]);
    wait_async();
    __syncthreads();

    // scores: this wave's 16 t-rows vs all 128 s-cols
    v8f sacc[8] = {};
#pragma unroll
    for (int ks2 = 0; ks2 < 2; ++ks2)
#pragma unroll
      for (int j = 0; j < 8; ++j) {
        v16bf b = load_frag(&Qs[j * 16][ks2 * 32], 72, lane);
        sacc[j] = wmma_bf16(ktf[ks2], b, sacc[j]);
      }

    // P~ = exp(score)/Z_s -> LDS (A-layout source for the PV WMMA)
#pragma unroll
    for (int j = 0; j < 8; ++j)
#pragma unroll
      for (int r = 0; r < 8; ++r) {
        int tr = w * 16 + ((lane >> 4) << 3) + r;
        int sc = j * 16 + (lane & 15);
        float p = __expf(sacc[j][r] * 0.125f) / Zs[sc];
        Ps[tr][sc] = (bf16)p;
      }
    __syncthreads();

    // oacc += P~ @ V   (K = 128 over s, 4 WMMA K-steps)
#pragma unroll
    for (int ks4 = 0; ks4 < 4; ++ks4) {
      v16bf a = load_frag(&Ps[w * 16][ks4 * 32], 136, lane);
#pragma unroll
      for (int j = 0; j < 4; ++j) {
        v16bf b = load_frag(&Vs[j * 16][ks4 * 32], 136, lane);
        oacc[j] = wmma_bf16(a, b, oacc[j]);
      }
    }
  }

#pragma unroll
  for (int j = 0; j < 4; ++j)
#pragma unroll
    for (int r = 0; r < 8; ++r) {
      int t = t0 + w * 16 + ((lane >> 4) << 3) + r;
      int e = j * 16 + (lane & 15);
      concat[(size_t)t * DMODEL + h * HD + e] = (bf16)oacc[j][r];
    }
}

// ---------------------------------------------------------------------------
// Dense GEMM wrappers (B pre-transposed, N-major [N][K])
// ---------------------------------------------------------------------------
__global__ void __launch_bounds__(256) k_gemm_proj(const bf16* A, const bf16* Bt,
                                                   const float* bias, bf16* C) {
  gemm_core<128, 128, 4, 2, false, true, false>(
      A, Bt, bias, C, DMODEL, DMODEL, DMODEL, DMODEL, blockIdx.x * 128,
      blockIdx.y * 128);
}
__global__ void __launch_bounds__(256) k_gemm_ff1(const bf16* A, const bf16* Bt,
                                                  const float* bias, bf16* C) {
  gemm_core<128, 128, 4, 2, true, true, false>(
      A, Bt, bias, C, DMODEL, DMODEL, DMODEL, FFDIM, blockIdx.x * 128,
      blockIdx.y * 128);
}
__global__ void __launch_bounds__(256) k_gemm_ff2(const bf16* A, const bf16* Bt,
                                                  const float* bias, float* C) {
  gemm_core<128, 128, 4, 2, false, false, false>(
      A, Bt, bias, C, FFDIM, FFDIM, FFDIM, DMODEL, blockIdx.x * 128,
      blockIdx.y * 128);
}

// ---------------------------------------------------------------------------
// Host launch
// ---------------------------------------------------------------------------
extern "C" void kernel_launch(void* const* d_in, const int* in_sizes, int n_in,
                              void* d_out, int out_size, void* d_ws,
                              size_t ws_size, hipStream_t stream) {
  const float* x  = (const float*)d_in[0];
  const float* Wq = (const float*)d_in[1];
  const float* bq = (const float*)d_in[2];
  const float* Wk = (const float*)d_in[3];
  const float* bk = (const float*)d_in[4];
  const float* Wv = (const float*)d_in[5];
  const float* bv = (const float*)d_in[6];
  const float* Wp = (const float*)d_in[7];
  const float* bp = (const float*)d_in[8];
  const float* W1 = (const float*)d_in[9];
  const float* b1 = (const float*)d_in[10];
  const float* W2 = (const float*)d_in[11];
  const float* b2 = (const float*)d_in[12];
  float* out = (float*)d_out;

  // workspace carve-up (256B aligned)
  char* p = (char*)d_ws;
  auto take = [&](size_t bytes) {
    char* q = p;
    p += (bytes + 255) & ~(size_t)255;
    return q;
  };
  bf16* xb   = (bf16*)take((size_t)SEQ * DMODEL * 2);
  bf16* Wqt  = (bf16*)take((size_t)NHEAD * DMODEL * HD * 2);  // [H][HD][D]
  bf16* Wkt  = (bf16*)take((size_t)NHEAD * DMODEL * HD * 2);
  bf16* Wvt  = (bf16*)take((size_t)NHEAD * DMODEL * HD * 2);
  bf16* Wpt  = (bf16*)take((size_t)DMODEL * DMODEL * 2);      // [N][K]
  bf16* W1t  = (bf16*)take((size_t)DMODEL * FFDIM * 2);       // [FF][D]
  bf16* W2t  = (bf16*)take((size_t)FFDIM * DMODEL * 2);       // [D][FF]
  bf16* Qb   = (bf16*)take((size_t)NHEAD * SEQ * HD * 2);     // [H][S][HD]
  bf16* Kb   = (bf16*)take((size_t)NHEAD * SEQ * HD * 2);
  bf16* Vt   = (bf16*)take((size_t)NHEAD * SEQ * HD * 2);     // [H][HD][S]
  float* Zb  = (float*)take((size_t)NHEAD * SEQ * 4);
  bf16* cat  = (bf16*)take((size_t)SEQ * DMODEL * 2);
  bf16* prj  = (bf16*)take((size_t)SEQ * DMODEL * 2);
  bf16* hid  = (bf16*)take((size_t)SEQ * FFDIM * 2);

  // x: plain convert; weights: transpose+convert (one-time, bandwidth-trivial)
  k_cvt_bf16<<<(SEQ * DMODEL + 1023) / 1024, 256, 0, stream>>>(x, xb,
                                                               SEQ * DMODEL);
  k_cvt_t<<<dim3(HD / 32, DMODEL / 32, NHEAD), 256, 0, stream>>>(Wq, Wqt,
                                                                 DMODEL, HD);
  k_cvt_t<<<dim3(HD / 32, DMODEL / 32, NHEAD), 256, 0, stream>>>(Wk, Wkt,
                                                                 DMODEL, HD);
  k_cvt_t<<<dim3(HD / 32, DMODEL / 32, NHEAD), 256, 0, stream>>>(Wv, Wvt,
                                                                 DMODEL, HD);
  k_cvt_t<<<dim3(DMODEL / 32, DMODEL / 32, 1), 256, 0, stream>>>(Wp, Wpt,
                                                                 DMODEL, DMODEL);
  k_cvt_t<<<dim3(FFDIM / 32, DMODEL / 32, 1), 256, 0, stream>>>(W1, W1t,
                                                                DMODEL, FFDIM);
  k_cvt_t<<<dim3(DMODEL / 32, FFDIM / 32, 1), 256, 0, stream>>>(W2, W2t,
                                                                FFDIM, DMODEL);

  k_zero_f32<<<(NHEAD * SEQ + 255) / 256, 256, 0, stream>>>(Zb, NHEAD * SEQ);

  // Q/K/V projections: [16 M-blocks] x [48 = 3 * 16 heads]
  k_qkv<<<dim3(SEQ / 128, 1, 48), 256, 0, stream>>>(xb, Wqt, Wkt, Wvt, bq, bk,
                                                    bv, Qb, Kb, Vt);
  // attention
  k_attn_z<<<dim3(SEQ / 128, 1, NHEAD), 256, 0, stream>>>(Qb, Kb, Zb);
  k_attn_out<<<dim3(SEQ / 128, 1, NHEAD), 256, 0, stream>>>(Qb, Kb, Vt, Zb, cat);
  // output projection + FFN
  k_gemm_proj<<<dim3(SEQ / 128, DMODEL / 128), 256, 0, stream>>>(cat, Wpt, bp, prj);
  k_gemm_ff1<<<dim3(SEQ / 128, FFDIM / 128), 256, 0, stream>>>(prj, W1t, b1, hid);
  k_gemm_ff2<<<dim3(SEQ / 128, DMODEL / 128), 256, 0, stream>>>(hid, W2t, b2, out);
}
